// GCN_3281355014802
// MI455X (gfx1250) — compile-verified, matrix operand order checked by reading, not run on hardware
//
#include <hip/hip_runtime.h>
#include <hip/hip_bf16.h>
#include <math.h>

#define N_FEAT    512
#define HIDDEN    16
#define N_CLASSES 40
#define N_CHUNKS  (N_FEAT / 32)    // 16 K-chunks of 32

typedef __attribute__((ext_vector_type(16))) __bf16 v16bf;
typedef __attribute__((ext_vector_type(8)))  float  v8f;

// ---------------------------------------------------------------------------
// deg[n] = 1 (self loop) + in-degree(n);  then dinv[n] = rsqrt(deg[n])
// ---------------------------------------------------------------------------
__global__ void k_deg_init(float* deg, int n) {
    int i = blockIdx.x * blockDim.x + threadIdx.x;
    if (i < n) deg[i] = 1.0f;
}

__global__ void k_deg_accum(const int* __restrict__ dst, float* deg, int nEdges) {
    int e = blockIdx.x * blockDim.x + threadIdx.x;
    if (e < nEdges) atomicAdd(&deg[dst[e]], 1.0f);
}

__global__ void k_rsqrt_inplace(float* deg, int n) {
    int i = blockIdx.x * blockDim.x + threadIdx.x;
    if (i < n) deg[i] = rsqrtf(deg[i]);   // deg >= 1 always (self loop)
}

// ---------------------------------------------------------------------------
// GEMM1: h1[N,16] = bf16(x[N,512]) @ bf16(W1[512,16]), f32 accumulate, WMMA.
// ONE 16-row M-tile per wave, NO tile loop (prevents LICM from hoisting the
// 16 B-fragments into registers and spilling them to scratch). K fully
// unrolled: 16 chunks of 32.
//
// A-frag (16x32 bf16, ISA 7.12.2): lane L holds row (L&15);
//   elems 0..7 -> K = kh..kh+7, elems 8..15 -> K = 16+kh..16+kh+7, kh=(L>>4)*8.
// B-frag (32x16 bf16): mirrored (lane L holds column L&15). W1 is pre-shuffled
//   into fragment order in LDS so a lane's fragment is one 32B contiguous,
//   32B-aligned block -> 2x ds_load_b128 per chunk, right next to its WMMA.
// C/D (16x16 f32): element v -> (M = v + 8*(L>>4), N = L&15).
// ---------------------------------------------------------------------------
__global__ void k_gemm1_wmma(const float* __restrict__ x,
                             const float* __restrict__ W1,
                             float* __restrict__ h1,
                             int nNodes, int numTiles) {
    // W1 in B-fragment order: [chunk][lane][16 elems], 8192 bf16 = 16 KB
    __shared__ __align__(32) __bf16 w1frag[N_CHUNKS * 32 * 16];
    for (int idx = threadIdx.x; idx < N_CHUNKS * 32 * 16; idx += blockDim.x) {
        int j     = idx & 15;          // element within fragment
        int ln    = (idx >> 4) & 31;   // owning lane
        int chunk = idx >> 9;          // K chunk
        int col   = ln & 15;
        int khf   = (ln >> 4) * 8;
        int k     = chunk * 32 + ((j < 8) ? (khf + j) : (16 + khf + (j - 8)));
        w1frag[idx] = (__bf16)W1[k * HIDDEN + col];
    }
    __syncthreads();

    const int lane = threadIdx.x & 31;
    const int wave = threadIdx.x >> 5;
    const int tile = blockIdx.x * (blockDim.x >> 5) + wave;
    if (tile >= numTiles) return;       // wave-uniform: EXEC all-ones below

    const int row = lane & 15;          // M row (A) / N column (B, C/D)
    const int kh  = (lane >> 4) * 8;    // K sub-range selector for this half-wave
    const v16bf* bfrags = (const v16bf*)w1frag;   // index: chunk*32 + lane

    // clamp load row so EXEC stays all-ones ahead of WMMA (ISA requirement)
    int ldRow = tile * 16 + row;
    if (ldRow > nNodes - 1) ldRow = nNodes - 1;
    const float* xrow = x + (size_t)ldRow * N_FEAT;

    // warm L2 for the back half of this row while the front half issues
    __builtin_prefetch(xrow + 256, 0, 1);   // global_prefetch_b8

    v8f acc = {};
#pragma unroll
    for (int kc = 0; kc < N_FEAT; kc += 32) {
        v16bf b = bfrags[(kc >> 5) * 32 + lane];   // 2x ds_load_b128
        v16bf a;
#pragma unroll
        for (int j = 0; j < 8; ++j) {              // 4x global_load_b128 + cvt_pk
            a[j]     = (__bf16)xrow[kc + kh + j];
            a[8 + j] = (__bf16)xrow[kc + 16 + kh + j];
        }
#if defined(__gfx1250__)
        acc = __builtin_amdgcn_wmma_f32_16x16x32_bf16(
            false, a, false, b, (short)0, acc, false, false);
#else
        // host-pass placeholder (never executed on device)
        for (int j = 0; j < 8; ++j) acc[j] += (float)a[j] * (float)b[j];
#endif
    }

    const int mbase = tile * 16 + ((lane >> 4) << 3);
#pragma unroll
    for (int v = 0; v < 8; ++v) {
        int m = mbase + v;
        if (m < nNodes) h1[(size_t)m * HIDDEN + row] = acc[v];
    }
}

// ---------------------------------------------------------------------------
// Aggregation (16 feats): g[n] = h[n]*dinv[n]^2  (self loop), then
// g[dst] += h[src]*dinv[src]*dinv[dst] over all edges.
// h/g buffers are 6.4 MB each -> resident in the 192 MB L2, so the random
// gathers and f32 atomics stay on-chip; only the edge list streams from HBM.
// ---------------------------------------------------------------------------
__global__ void k_self_init16(const float* __restrict__ h,
                              const float* __restrict__ dinv,
                              float* __restrict__ g, int total) {
    int i = blockIdx.x * blockDim.x + threadIdx.x;
    if (i < total) {
        float d = dinv[i >> 4];
        g[i] = h[i] * d * d;
    }
}

__global__ void k_edge_scatter16(const int* __restrict__ src,
                                 const int* __restrict__ dst,
                                 const float* __restrict__ h,
                                 const float* __restrict__ dinv,
                                 float* __restrict__ g, int total) {
    int i = blockIdx.x * blockDim.x + threadIdx.x;   // = edge*16 + feat
    if (i < total) {
        int e = i >> 4, f = i & 15;
        int s = src[e], d = dst[e];
        float norm = dinv[s] * dinv[d];
        atomicAdd(&g[(size_t)d * HIDDEN + f], h[(size_t)s * HIDDEN + f] * norm);
    }
}

__global__ void k_bias_relu16(float* __restrict__ g,
                              const float* __restrict__ b1, int total) {
    int i = blockIdx.x * blockDim.x + threadIdx.x;
    if (i < total) g[i] = fmaxf(g[i] + b1[i & 15], 0.0f);
}

// ---------------------------------------------------------------------------
// Fused GEMM2 (16 -> 40) + bias + log_softmax. One thread per node;
// W2/b2 staged in LDS (2.7 KB). 640 FMA + softmax over 40 per thread.
// ---------------------------------------------------------------------------
__global__ void k_out_logsoftmax(const float* __restrict__ g2,
                                 const float* __restrict__ W2,
                                 const float* __restrict__ b2,
                                 float* __restrict__ out, int nNodes) {
    __shared__ float w2s[HIDDEN * N_CLASSES];
    __shared__ float b2s[N_CLASSES];
    for (int i = threadIdx.x; i < HIDDEN * N_CLASSES; i += blockDim.x) w2s[i] = W2[i];
    for (int i = threadIdx.x; i < N_CLASSES; i += blockDim.x) b2s[i] = b2[i];
    __syncthreads();

    int n = blockIdx.x * blockDim.x + threadIdx.x;
    if (n >= nNodes) return;

    float gk[HIDDEN];
#pragma unroll
    for (int k = 0; k < HIDDEN; ++k) gk[k] = g2[(size_t)n * HIDDEN + k];

    float z[N_CLASSES];
    float mx = -INFINITY;
#pragma unroll
    for (int c = 0; c < N_CLASSES; ++c) {
        float t = b2s[c];
#pragma unroll
        for (int k = 0; k < HIDDEN; ++k) t = fmaf(gk[k], w2s[k * N_CLASSES + c], t);
        z[c] = t;
        mx = fmaxf(mx, t);
    }
    float ssum = 0.0f;
#pragma unroll
    for (int c = 0; c < N_CLASSES; ++c) ssum += __expf(z[c] - mx);
    float lse = mx + __logf(ssum);
#pragma unroll
    for (int c = 0; c < N_CLASSES; ++c) out[(size_t)n * N_CLASSES + c] = z[c] - lse;
}

// ---------------------------------------------------------------------------
extern "C" void kernel_launch(void* const* d_in, const int* in_sizes, int n_in,
                              void* d_out, int out_size, void* d_ws, size_t ws_size,
                              hipStream_t stream) {
    const float* x  = (const float*)d_in[0];
    const int*   ei = (const int*)d_in[1];   // edge_index [2, E] (int per harness)
    const float* W1 = (const float*)d_in[2];
    const float* b1 = (const float*)d_in[3];
    const float* W2 = (const float*)d_in[4];
    const float* b2 = (const float*)d_in[5];
    float*       out = (float*)d_out;

    const int N = in_sizes[0] / N_FEAT;       // 100000
    const int E = in_sizes[1] / 2;            // 3200000
    const int* srcIdx = ei;
    const int* dstIdx = ei + E;

    // workspace layout (floats): dinv[N] | h1[N*16] | g1[N*16] | g2[N*16]
    float* dinv = (float*)d_ws;
    float* h1 = dinv + N;
    float* g1 = h1 + (size_t)N * HIDDEN;
    float* g2 = g1 + (size_t)N * HIDDEN;

    const int B = 256;
    const int numTiles = (N + 15) / 16;
    const int nodeFeat = N * HIDDEN;
    const int edgeFeat = E * HIDDEN;          // 51.2M, fits int

    // 1) degrees -> dinv
    k_deg_init<<<(N + B - 1) / B, B, 0, stream>>>(dinv, N);
    k_deg_accum<<<(E + B - 1) / B, B, 0, stream>>>(dstIdx, dinv, E);
    k_rsqrt_inplace<<<(N + B - 1) / B, B, 0, stream>>>(dinv, N);

    // 2) h1 = x @ W1 via WMMA (8 waves/block, exactly one 16-row tile per wave)
    k_gemm1_wmma<<<(numTiles + 7) / 8, B, 0, stream>>>(x, W1, h1, N, numTiles);

    // 3) layer-1 aggregation + bias + relu  (a1 stored in-place in g1)
    k_self_init16<<<(nodeFeat + B - 1) / B, B, 0, stream>>>(h1, dinv, g1, nodeFeat);
    k_edge_scatter16<<<(edgeFeat + B - 1) / B, B, 0, stream>>>(srcIdx, dstIdx, h1, dinv, g1, edgeFeat);
    k_bias_relu16<<<(nodeFeat + B - 1) / B, B, 0, stream>>>(g1, b1, nodeFeat);

    // 4) layer-2 aggregation FIRST (agg commutes with @W2) -> g2 (16-wide)
    k_self_init16<<<(nodeFeat + B - 1) / B, B, 0, stream>>>(g1, dinv, g2, nodeFeat);
    k_edge_scatter16<<<(edgeFeat + B - 1) / B, B, 0, stream>>>(srcIdx, dstIdx, g1, dinv, g2, edgeFeat);

    // 5) fused (g2 @ W2 + b2) + log_softmax
    k_out_logsoftmax<<<(N + B - 1) / B, B, 0, stream>>>(g2, W2, b2, out, N);
}
